// Frequency2SpatialLayer_55808805044896
// MI455X (gfx1250) — matrix operand order repaired and checked by reading, not compiled, not run on hardware
//
#include <hip/hip_runtime.h>
#include <hip/hip_bf16.h>

// MI455X / gfx1250: wave32, WMMA bf16 16x16x32, f32 accumulate.
// DFT-as-GEMM with hi/lo bf16 split for ~fp32 accuracy.
// A-operand split is hoisted out of the GEMM loop into a memory-bound pre-pass
// so the GEMM inner loop is pure loads + v_wmma (no VALU conversion storm).

typedef __attribute__((ext_vector_type(16))) __bf16 v16bf;
typedef __attribute__((ext_vector_type(8)))  __bf16 v8bf;
typedef __attribute__((ext_vector_type(8)))  float  v8f;

#define DFT_N   1024
#define NROWS   16384                    // 16 batches * 1024 projections
#define TBL_ELT (1u << 20)               // 1024*1024 entries per basis table
#define A_ELT   ((size_t)NROWS * DFT_N)  // 16M elements per split-A array

static __device__ __forceinline__ __bf16 bf16_from_bits(unsigned short v) {
    __bf16 r; __builtin_memcpy(&r, &v, 2); return r;
}
static __device__ __forceinline__ unsigned short bf16_bits_rn(float f) {
    __bf16 h = (__bf16)f; unsigned short b; __builtin_memcpy(&b, &h, 2); return b;
}
// Monotone float -> u32 key (total order), so u32 atomicMin/Max == float min/max.
static __device__ __forceinline__ unsigned fkey(float f) {
    unsigned u = __float_as_uint(f);
    return (u & 0x80000000u) ? ~u : (u | 0x80000000u);
}
static __device__ __forceinline__ float fdecode(unsigned k) {
    unsigned u = (k & 0x80000000u) ? (k ^ 0x80000000u) : ~k;
    return __uint_as_float(u);
}
static __device__ __forceinline__ v16bf cat8(v8bf a, v8bf b) {
    return __builtin_shufflevector(a, b, 0,1,2,3,4,5,6,7,8,9,10,11,12,13,14,15);
}

__global__ void init_minmax(unsigned* wsu) {
    if (threadIdx.x == 0) { wsu[0] = 0xFFFFFFFFu; wsu[1] = 0u; }
}

// Transposed DFT basis tables T[n][k], bf16 hi/lo split.
//   Bc[k][n] = (-1)^n/N *  cos(2*pi*k*n/N)
//   Bs[k][n] = (-1)^n/N * -sin(2*pi*k*n/N)
__global__ void build_tables(__bf16* __restrict__ bc_hi, __bf16* __restrict__ bc_lo,
                             __bf16* __restrict__ bs_hi, __bf16* __restrict__ bs_lo) {
    unsigned tid = blockIdx.x * blockDim.x + threadIdx.x;   // [n][k] flat
    unsigned n = tid >> 10, k = tid & 1023u;
    unsigned ph = (k * n) & 1023u;                          // k*n mod N (exact)
    float ang = (float)ph * 6.135923151542565e-03f;         // 2*pi/1024
    float s, c;
    sincosf(ang, &s, &c);
    float sc = (n & 1u) ? -9.765625e-04f : 9.765625e-04f;   // +-1/N
    float bc = c * sc;
    float bs = -s * sc;
    unsigned ub = __float_as_uint(bc);
    unsigned us = __float_as_uint(bs);
    bc_hi[tid] = bf16_from_bits((unsigned short)(ub >> 16));          // truncated hi
    bc_lo[tid] = (__bf16)(bc - __uint_as_float(ub & 0xFFFF0000u));    // exact residual -> RN
    bs_hi[tid] = bf16_from_bits((unsigned short)(us >> 16));
    bs_lo[tid] = (__bf16)(bs - __uint_as_float(us & 0xFFFF0000u));
}

// Pre-pass: de-interleave x[b][c][p][d] and split f32 -> bf16 hi/lo.
// Output arrays are [row=b*1024+p][d] row-major bf16. 4 f32 per thread.
__global__ void split_a(const float* __restrict__ x,
                        __bf16* __restrict__ reH, __bf16* __restrict__ reL,
                        __bf16* __restrict__ imH, __bf16* __restrict__ imL) {
    size_t idx = ((size_t)blockIdx.x * blockDim.x + threadIdx.x) * 4;  // flat f32 index
    unsigned b = (unsigned)(idx >> 21);
    unsigned c = (unsigned)(idx >> 20) & 1u;
    unsigned p = (unsigned)(idx >> 10) & 1023u;
    unsigned d = (unsigned)idx & 1023u;
    size_t out = (((size_t)b << 10) | p) * DFT_N + d;

    float4 v = *reinterpret_cast<const float4*>(x + idx);
    float f[4] = {v.x, v.y, v.z, v.w};
    unsigned h[4], l[4];
#pragma unroll
    for (int e = 0; e < 4; ++e) {
        unsigned u = __float_as_uint(f[e]);
        h[e] = u >> 16;                                                    // truncated hi
        l[e] = bf16_bits_rn(f[e] - __uint_as_float(u & 0xFFFF0000u));      // residual, RN
    }
    uint2 hv = make_uint2(h[0] | (h[1] << 16), h[2] | (h[3] << 16));
    uint2 lv = make_uint2(l[0] | (l[1] << 16), l[2] | (l[3] << 16));
    __bf16* dstH = c ? imH : reH;
    __bf16* dstL = c ? imL : reL;
    *reinterpret_cast<uint2*>(dstH + out) = hv;
    *reinterpret_cast<uint2*>(dstL + out) = lv;
}

// ---- shared epilogue: store C frag, global min/max via sortable-u32 atomics ----
static __device__ __forceinline__ void tile_epilogue(
        const v8f& acc, int m_tile, int ncol, int lane, int wloc,
        float* __restrict__ y, unsigned* __restrict__ wsu,
        unsigned* smn, unsigned* smx) {
    const int row0 = (m_tile << 4) + ((lane >= 16) ? 8 : 0);
    unsigned kmn = 0xFFFFFFFFu, kmx = 0u;
#pragma unroll
    for (int v = 0; v < 8; ++v) {
        float val = acc[v];
        y[(size_t)(row0 + v) * DFT_N + ncol] = val;
        unsigned kk = fkey(val);
        kmn = (kk < kmn) ? kk : kmn;
        kmx = (kk > kmx) ? kk : kmx;
    }
#pragma unroll
    for (int off = 16; off > 0; off >>= 1) {
        unsigned a = (unsigned)__shfl_xor((int)kmn, off, 32);
        unsigned b = (unsigned)__shfl_xor((int)kmx, off, 32);
        kmn = (a < kmn) ? a : kmn;
        kmx = (b > kmx) ? b : kmx;
    }
    if (lane == 0) { smn[wloc] = kmn; smx[wloc] = kmx; }
    __syncthreads();
    if (threadIdx.x == 0) {
        unsigned m1 = smn[0], m2 = smx[0];
        for (int i = 1; i < 8; ++i) {
            m1 = (smn[i] < m1) ? smn[i] : m1;
            m2 = (smx[i] > m2) ? smx[i] : m2;
        }
        atomicMin(&wsu[0], m1);
        atomicMax(&wsu[1], m2);
    }
}

// Fast path: A pre-split in ws. Inner loop = 16x b128 loads + 6x v_wmma.
__global__ void __launch_bounds__(256)
dft_wmma_pre(const __bf16* __restrict__ reH, const __bf16* __restrict__ reL,
             const __bf16* __restrict__ imH, const __bf16* __restrict__ imL,
             float* __restrict__ y,
             const __bf16* __restrict__ bc_hi, const __bf16* __restrict__ bc_lo,
             const __bf16* __restrict__ bs_hi, const __bf16* __restrict__ bs_lo,
             unsigned* __restrict__ wsu) {
    const int lane = threadIdx.x & 31;
    const int wloc = threadIdx.x >> 5;
    const int wid  = blockIdx.x * 8 + wloc;        // 0..65535
    const int n_tile = wid >> 10;                  // 0..63 (same for whole block -> B reuse)
    const int m_tile = wid & 1023;                 // 0..1023

    // A frag: lane row M = m_tile*16 + lane%16; K = e + klo
    const int    mrow = (m_tile << 4) + (lane & 15);
    const size_t abase = ((size_t)mrow << 10);
    const int    klo   = (lane >= 16) ? 16 : 0;
    // B frag: K = e + koffB, N = lane%16 (transposed tables: contiguous)
    const int    koffB = (lane >= 16) ? 16 : 0;
    const int    ncol  = (n_tile << 4) + (lane & 15);
    const size_t brow  = ((size_t)ncol << 10) + (size_t)koffB;

    v8f acc = {0.f, 0.f, 0.f, 0.f, 0.f, 0.f, 0.f, 0.f};

    for (int kb = 0; kb < DFT_N; kb += 32) {
        const size_t a0 = abase + kb + klo;         // cols [klo..+7]
        const size_t a1 = a0 + 8;                   // cols [8+klo..+7]
        v16bf aReH = cat8(*(const v8bf*)(reH + a0), *(const v8bf*)(reH + a1));
        v16bf aReL = cat8(*(const v8bf*)(reL + a0), *(const v8bf*)(reL + a1));
        v16bf aImH = cat8(*(const v8bf*)(imH + a0), *(const v8bf*)(imH + a1));
        v16bf aImL = cat8(*(const v8bf*)(imL + a0), *(const v8bf*)(imL + a1));
        const v16bf bCH = *reinterpret_cast<const v16bf*>(bc_hi + brow + kb);
        const v16bf bCL = *reinterpret_cast<const v16bf*>(bc_lo + brow + kb);
        const v16bf bSH = *reinterpret_cast<const v16bf*>(bs_hi + brow + kb);
        const v16bf bSL = *reinterpret_cast<const v16bf*>(bs_lo + brow + kb);

        // D += Ah*Bh + Ah*Bl + Al*Bh (drop lo*lo), for Re*Bc and Im*Bs
        acc = __builtin_amdgcn_wmma_f32_16x16x32_bf16(false, aReH, false, bCH, (short)0, acc, false, false);
        acc = __builtin_amdgcn_wmma_f32_16x16x32_bf16(false, aReH, false, bCL, (short)0, acc, false, false);
        acc = __builtin_amdgcn_wmma_f32_16x16x32_bf16(false, aReL, false, bCH, (short)0, acc, false, false);
        acc = __builtin_amdgcn_wmma_f32_16x16x32_bf16(false, aImH, false, bSH, (short)0, acc, false, false);
        acc = __builtin_amdgcn_wmma_f32_16x16x32_bf16(false, aImH, false, bSL, (short)0, acc, false, false);
        acc = __builtin_amdgcn_wmma_f32_16x16x32_bf16(false, aImL, false, bSH, (short)0, acc, false, false);
    }

    __shared__ unsigned smn[8], smx[8];
    tile_epilogue(acc, m_tile, ncol, lane, wloc, y, wsu, smn, smx);
}

// Fallback path (small ws): split A on the fly inside the loop.
__global__ void __launch_bounds__(256)
dft_wmma_fused(const float* __restrict__ x, float* __restrict__ y,
               const __bf16* __restrict__ bc_hi, const __bf16* __restrict__ bc_lo,
               const __bf16* __restrict__ bs_hi, const __bf16* __restrict__ bs_lo,
               unsigned* __restrict__ wsu) {
    const int lane = threadIdx.x & 31;
    const int wloc = threadIdx.x >> 5;
    const int wid  = blockIdx.x * 8 + wloc;
    const int n_tile = wid >> 10;
    const int m_tile = wid & 1023;

    const int  mrow  = (m_tile << 4) + (lane & 15);
    const int  batch = mrow >> 10;
    const int  proj  = mrow & 1023;
    const float* rowRe = x + ((size_t)batch << 21) + ((size_t)proj << 10);
    const float* rowIm = rowRe + (1u << 20);

    const int klo   = (lane >= 16) ? 16 : 0;
    const int koffB = (lane >= 16) ? 16 : 0;
    const int ncol  = (n_tile << 4) + (lane & 15);
    const size_t brow = ((size_t)ncol << 10) + (size_t)koffB;

    v8f acc = {0.f, 0.f, 0.f, 0.f, 0.f, 0.f, 0.f, 0.f};

    for (int kb = 0; kb < DFT_N; kb += 32) {
        const float4* pre = reinterpret_cast<const float4*>(rowRe + kb + klo);
        const float4* pim = reinterpret_cast<const float4*>(rowIm + kb + klo);
        float4 r0 = pre[0], r1 = pre[1], r2 = pre[2], r3 = pre[3];
        float4 i0 = pim[0], i1 = pim[1], i2 = pim[2], i3 = pim[3];
        float fr[16] = {r0.x,r0.y,r0.z,r0.w, r1.x,r1.y,r1.z,r1.w,
                        r2.x,r2.y,r2.z,r2.w, r3.x,r3.y,r3.z,r3.w};
        float fi[16] = {i0.x,i0.y,i0.z,i0.w, i1.x,i1.y,i1.z,i1.w,
                        i2.x,i2.y,i2.z,i2.w, i3.x,i3.y,i3.z,i3.w};
        v16bf aReH, aReL, aImH, aImL;
#pragma unroll
        for (int e = 0; e < 16; ++e) {
            unsigned ur = __float_as_uint(fr[e]);
            aReH[e] = bf16_from_bits((unsigned short)(ur >> 16));
            aReL[e] = (__bf16)(fr[e] - __uint_as_float(ur & 0xFFFF0000u));
            unsigned ui = __float_as_uint(fi[e]);
            aImH[e] = bf16_from_bits((unsigned short)(ui >> 16));
            aImL[e] = (__bf16)(fi[e] - __uint_as_float(ui & 0xFFFF0000u));
        }
        const v16bf bCH = *reinterpret_cast<const v16bf*>(bc_hi + brow + kb);
        const v16bf bCL = *reinterpret_cast<const v16bf*>(bc_lo + brow + kb);
        const v16bf bSH = *reinterpret_cast<const v16bf*>(bs_hi + brow + kb);
        const v16bf bSL = *reinterpret_cast<const v16bf*>(bs_lo + brow + kb);

        acc = __builtin_amdgcn_wmma_f32_16x16x32_bf16(false, aReH, false, bCH, (short)0, acc, false, false);
        acc = __builtin_amdgcn_wmma_f32_16x16x32_bf16(false, aReH, false, bCL, (short)0, acc, false, false);
        acc = __builtin_amdgcn_wmma_f32_16x16x32_bf16(false, aReL, false, bCH, (short)0, acc, false, false);
        acc = __builtin_amdgcn_wmma_f32_16x16x32_bf16(false, aImH, false, bSH, (short)0, acc, false, false);
        acc = __builtin_amdgcn_wmma_f32_16x16x32_bf16(false, aImH, false, bSL, (short)0, acc, false, false);
        acc = __builtin_amdgcn_wmma_f32_16x16x32_bf16(false, aImL, false, bSH, (short)0, acc, false, false);
    }

    __shared__ unsigned smn[8], smx[8];
    tile_epilogue(acc, m_tile, ncol, lane, wloc, y, wsu, smn, smx);
}

__global__ void normalize_k(float* __restrict__ y, const unsigned* __restrict__ wsu) {
    float mn = fdecode(wsu[0]);
    float mx = fdecode(wsu[1]);
    float sc = 1.0f / (mx - mn);
    float4* p = reinterpret_cast<float4*>(y);
    size_t stride = (size_t)gridDim.x * blockDim.x;
    for (size_t j = (size_t)blockIdx.x * blockDim.x + threadIdx.x;
         j < (size_t)NROWS * DFT_N / 4; j += stride) {
        float4 v = p[j];
        v.x = (v.x - mn) * sc; v.y = (v.y - mn) * sc;
        v.z = (v.z - mn) * sc; v.w = (v.w - mn) * sc;
        p[j] = v;
    }
}

extern "C" void kernel_launch(void* const* d_in, const int* in_sizes, int n_in,
                              void* d_out, int out_size, void* d_ws, size_t ws_size,
                              hipStream_t stream) {
    (void)in_sizes; (void)n_in; (void)out_size;
    const float* x = (const float*)d_in[0];          // [16,2,1024,1024] f32
    float* y = (float*)d_out;                        // [16,1,1024,1024] f32

    // ws layout: [0..7] min/max keys; +64B: 4 bf16 basis tables (8 MB);
    // then (fast path) 4 bf16 split-A arrays (128 MB).
    unsigned* wsu = (unsigned*)d_ws;
    __bf16* bch = (__bf16*)((char*)d_ws + 64);
    __bf16* bcl = bch + TBL_ELT;
    __bf16* bsh = bcl + TBL_ELT;
    __bf16* bsl = bsh + TBL_ELT;
    __bf16* reH = bsl + TBL_ELT;
    __bf16* reL = reH + A_ELT;
    __bf16* imH = reL + A_ELT;
    __bf16* imL = imH + A_ELT;
    const size_t need_fast = 64 + 4 * (size_t)TBL_ELT * 2 + 4 * A_ELT * 2;

    init_minmax<<<1, 32, 0, stream>>>(wsu);
    build_tables<<<TBL_ELT / 256, 256, 0, stream>>>(bch, bcl, bsh, bsl);
    const int gemm_blocks = (NROWS / 16) * (DFT_N / 16) / 8;
    if (ws_size >= need_fast) {
        split_a<<<(2u * TBL_ELT * 16 / 4) / 256, 256, 0, stream>>>(x, reH, reL, imH, imL);
        dft_wmma_pre<<<gemm_blocks, 256, 0, stream>>>(reH, reL, imH, imL, y,
                                                      bch, bcl, bsh, bsl, wsu);
    } else {
        dft_wmma_fused<<<gemm_blocks, 256, 0, stream>>>(x, y, bch, bcl, bsh, bsl, wsu);
    }
    normalize_k<<<4096, 256, 0, stream>>>(y, wsu);
}
